// HypCoAttn_52913997086888
// MI455X (gfx1250) — compile-verified, matrix operand order checked by reading, not run on hardware
//
#include <hip/hip_runtime.h>
#include <hip/hip_bf16.h>
#include <math.h>

// ---------------------------------------------------------------------------
// Hyperbolic (Lorentz) attention for MI455X / gfx1250.
//   logits = -( q_flip . k ) / scale + bias      (q_flip: q0 negated)
//   att    = softmax(logits)
//   ave    = att @ V
//   out    = ave / sqrt(clip(|2*ave0^2 - sum(ave^2)|, eps))
// Flash-attention structure, bf16 WMMA (v_wmma_f32_16x16x32_bf16), f32 acc.
// ---------------------------------------------------------------------------

typedef __attribute__((ext_vector_type(16))) __bf16 bf16x16;
typedef __attribute__((ext_vector_type(8)))  float  f32x8;

#define BATCH   8
#define NQ      2048
#define NK      2048
#define DIM     512
#define BQ      16          // queries per workgroup
#define KT      128         // keys per k-iteration
#define NWAVES  8
#define EPS_L   1e-8f

#define SMEM_BYTES ((BQ*DIM + KT*DIM + DIM*KT + BQ*KT) * 2 + (3*128 + 5*BQ) * 4)

__device__ __forceinline__ unsigned short f2bf(float f) {
    unsigned u = __float_as_uint(f);
    unsigned r = u + 0x7FFFu + ((u >> 16) & 1u);   // round-to-nearest-even
    return (unsigned short)(r >> 16);
}

__device__ __forceinline__ unsigned long long pack4bf(float4 x) {
    return (unsigned long long)f2bf(x.x)
         | ((unsigned long long)f2bf(x.y) << 16)
         | ((unsigned long long)f2bf(x.z) << 32)
         | ((unsigned long long)f2bf(x.w) << 48);
}

__global__ void __launch_bounds__(256, 1)
hyp_attn_kernel(const float* __restrict__ qs, const float* __restrict__ ks,
                const float* __restrict__ vs, const float* __restrict__ scale,
                const float* __restrict__ bias, float* __restrict__ out)
{
    extern __shared__ char smem[];
    unsigned short* Qs  = (unsigned short*)smem;   // [BQ][DIM]  bf16, sign/scale folded
    unsigned short* Ksm = Qs  + BQ * DIM;          // [KT][DIM]  bf16
    unsigned short* Vts = Ksm + KT * DIM;          // [DIM][KT]  bf16 (transposed V tile)
    unsigned short* Ps  = Vts + DIM * KT;          // [BQ][KT]   bf16 (exp'd probabilities)
    float* pmax  = (float*)(Ps + BQ * KT);         // [NWAVES][16] partial row max
    float* psum  = pmax + 128;                     // [NWAVES][16] partial row sum
    float* pssq  = psum + 128;                     // [NWAVES][16] partial sum-of-squares
    float* rowM  = pssq + 128;                     // [16] running max
    float* rowL  = rowM + BQ;                      // [16] running denom
    float* rowS  = rowL + BQ;                      // [16] rescale factor
    float* rowA0 = rowS + BQ;                      // [16] ave[., 0]
    float* rowD  = rowA0 + BQ;                     // [16] 1/denom (Lorentz)

    const int tid  = threadIdx.x;
    const int wave = tid >> 5;
    const int lane = tid & 31;
    const int half = lane >> 4;        // 0: lanes 0-15, 1: lanes 16-31
    const int ln   = lane & 15;        // column/lane index inside half

    const int b  = blockIdx.x / (NQ / BQ);
    const int q0 = (blockIdx.x % (NQ / BQ)) * BQ;

    const float inv_scale = 1.0f / scale[0];
    const float biasv     = bias[0];

    if (tid < BQ) { rowM[tid] = -1e30f; rowL[tid] = 0.0f; }

    // ---- Stage Q once: Qs[m][d] = (d==0 ? +q : -q) * inv_scale, bf16 ----
    {
        const float* qb = qs + ((size_t)b * NQ + q0) * DIM;
        #pragma unroll
        for (int i = 0; i < 8; ++i) {
            int idx = (tid * 8 + i) * 4;           // 0 .. 8188, step 4
            int r = idx >> 9;                      // / DIM
            int c = idx & (DIM - 1);
            float4 v = *(const float4*)(qb + (size_t)r * DIM + c);
            float s0 = (c == 0) ? v.x : -v.x;      // -q_flip
            Qs[r * DIM + c + 0] = f2bf(s0   * inv_scale);
            Qs[r * DIM + c + 1] = f2bf(-v.y * inv_scale);
            Qs[r * DIM + c + 2] = f2bf(-v.z * inv_scale);
            Qs[r * DIM + c + 3] = f2bf(-v.w * inv_scale);
        }
    }

    f32x8 acc[4];
    #pragma unroll
    for (int t = 0; t < 4; ++t) acc[t] = {};

    const float* kb = ks + (size_t)b * NK * DIM;
    const float* vb = vs + (size_t)b * NK * DIM;

    for (int k0 = 0; k0 < NK; k0 += KT) {
        __syncthreads();   // previous iteration's readers of Ksm/Vts/Ps done

        // ---- Stage K tile (row-major bf16) and V tile (transposed bf16) ----
        {
            int row = tid >> 1, seg = tid & 1;     // 2 threads / key row
            const float* kr = kb + (size_t)(k0 + row) * DIM + seg * 256;
            const float* vr = vb + (size_t)(k0 + row) * DIM + seg * 256;
            unsigned short* kd = Ksm + row * DIM + seg * 256;
            #pragma unroll 4
            for (int j = 0; j < 64; ++j) {
                float4 x = *(const float4*)(kr + j * 4);
                *(unsigned long long*)(kd + j * 4) = pack4bf(x);
                float4 y = *(const float4*)(vr + j * 4);
                int dbase = seg * 256 + j * 4;
                Vts[(dbase + 0) * KT + row] = f2bf(y.x);
                Vts[(dbase + 1) * KT + row] = f2bf(y.y);
                Vts[(dbase + 2) * KT + row] = f2bf(y.z);
                Vts[(dbase + 3) * KT + row] = f2bf(y.w);
            }
        }
        if (k0 + KT < NK) {   // global_prefetch_b8 for the next tile
            __builtin_prefetch(kb + (size_t)(k0 + KT) * DIM + (size_t)tid * 256, 0, 1);
            __builtin_prefetch(vb + (size_t)(k0 + KT) * DIM + (size_t)tid * 256, 0, 1);
        }
        __syncthreads();

        // ---- S = Q . K^T : each wave one 16x16 tile (cols wave*16..+16) ----
        f32x8 s = {};
        #pragma unroll
        for (int dk = 0; dk < DIM / 32; ++dk) {
            const bf16x16 a  = *(const bf16x16*)(Qs  + ln * DIM + dk * 32 + half * 16);
            const bf16x16 bm = *(const bf16x16*)(Ksm + (wave * 16 + ln) * DIM + dk * 32 + half * 16);
            s = __builtin_amdgcn_wmma_f32_16x16x32_bf16(false, a, false, bm,
                                                        (short)0, s, false, false);
        }
        #pragma unroll
        for (int r = 0; r < 8; ++r) s[r] += biasv;

        // ---- online softmax: per-row max (cross-lane within 16-group) ----
        #pragma unroll
        for (int r = 0; r < 8; ++r) {
            float v = s[r];
            v = fmaxf(v, __shfl_xor(v, 1, 32));
            v = fmaxf(v, __shfl_xor(v, 2, 32));
            v = fmaxf(v, __shfl_xor(v, 4, 32));
            v = fmaxf(v, __shfl_xor(v, 8, 32));
            if (ln == 0) pmax[wave * 16 + r + half * 8] = v;
        }
        __syncthreads();
        if (tid < BQ) {
            float nm = rowM[tid];
            #pragma unroll
            for (int w = 0; w < NWAVES; ++w) nm = fmaxf(nm, pmax[w * 16 + tid]);
            rowS[tid] = __expf(rowM[tid] - nm);
            rowM[tid] = nm;
        }
        __syncthreads();

        // ---- P = exp(s - rowmax), write bf16 to LDS, partial row sums ----
        #pragma unroll
        for (int r = 0; r < 8; ++r) {
            float nm = rowM[r + half * 8];
            float p  = __expf(s[r] - nm);
            Ps[(r + half * 8) * KT + wave * 16 + ln] = f2bf(p);
            float v = p;
            v += __shfl_xor(v, 1, 32);
            v += __shfl_xor(v, 2, 32);
            v += __shfl_xor(v, 4, 32);
            v += __shfl_xor(v, 8, 32);
            if (ln == 0) psum[wave * 16 + r + half * 8] = v;
        }
        __syncthreads();
        if (tid < BQ) {
            float l = rowL[tid] * rowS[tid];
            #pragma unroll
            for (int w = 0; w < NWAVES; ++w) l += psum[w * 16 + tid];
            rowL[tid] = l;
        }

        // ---- O = diag(rescale)*O + P . V : wave owns 64 D-columns ----
        float rs[8];
        #pragma unroll
        for (int r = 0; r < 8; ++r) rs[r] = rowS[r + half * 8];
        #pragma unroll
        for (int t = 0; t < 4; ++t) {
            #pragma unroll
            for (int r = 0; r < 8; ++r) acc[t][r] *= rs[r];
            const int dbase = wave * 64 + t * 16;
            #pragma unroll
            for (int kk = 0; kk < KT / 32; ++kk) {
                const bf16x16 a  = *(const bf16x16*)(Ps  + ln * KT + kk * 32 + half * 16);
                const bf16x16 bm = *(const bf16x16*)(Vts + (dbase + ln) * KT + kk * 32 + half * 16);
                acc[t] = __builtin_amdgcn_wmma_f32_16x16x32_bf16(false, a, false, bm,
                                                                 (short)0, acc[t], false, false);
            }
        }
    }

    // ---- Epilogue: ave = O / L, then Lorentz renormalization ----
    __syncthreads();
    float il[8];
    #pragma unroll
    for (int r = 0; r < 8; ++r) il[r] = 1.0f / fmaxf(rowL[r + half * 8], 1e-30f);

    #pragma unroll
    for (int r = 0; r < 8; ++r) {
        float ssq = 0.0f;
        #pragma unroll
        for (int t = 0; t < 4; ++t) {
            float av = acc[t][r] * il[r];
            ssq += av * av;
        }
        ssq += __shfl_xor(ssq, 1, 32);
        ssq += __shfl_xor(ssq, 2, 32);
        ssq += __shfl_xor(ssq, 4, 32);
        ssq += __shfl_xor(ssq, 8, 32);
        if (ln == 0) pssq[wave * 16 + r + half * 8] = ssq;
        if (wave == 0 && ln == 0)   // element d==0 lives in wave 0, tile 0, col 0
            rowA0[r + half * 8] = acc[0][r] * il[r];
    }
    __syncthreads();
    if (tid < BQ) {
        float ssq = 0.0f;
        #pragma unroll
        for (int w = 0; w < NWAVES; ++w) ssq += pssq[w * 16 + tid];
        float a0  = rowA0[tid];
        float lor = 2.0f * a0 * a0 - ssq;      // -<ave,ave>_L
        rowD[tid] = 1.0f / sqrtf(fmaxf(fabsf(lor), EPS_L));
    }
    __syncthreads();

    float rd[8];
    #pragma unroll
    for (int r = 0; r < 8; ++r) rd[r] = rowD[r + half * 8];

    float* ob = out + ((size_t)b * NQ + q0) * DIM;
    #pragma unroll
    for (int t = 0; t < 4; ++t) {
        const int dcol = wave * 64 + t * 16 + ln;
        #pragma unroll
        for (int r = 0; r < 8; ++r) {
            const int m = r + half * 8;
            ob[(size_t)m * DIM + dcol] = acc[t][r] * il[r] * rd[r];
        }
    }
}

extern "C" void kernel_launch(void* const* d_in, const int* in_sizes, int n_in,
                              void* d_out, int out_size, void* d_ws, size_t ws_size,
                              hipStream_t stream) {
    const float* qs    = (const float*)d_in[0];
    const float* ksp   = (const float*)d_in[1];
    const float* vsp   = (const float*)d_in[2];
    const float* scale = (const float*)d_in[3];
    const float* bias  = (const float*)d_in[4];
    float* out = (float*)d_out;
    (void)in_sizes; (void)n_in; (void)out_size; (void)d_ws; (void)ws_size;

    dim3 grid(BATCH * (NQ / BQ));   // 1024 workgroups
    dim3 block(256);                // 8 wave32s
    hipLaunchKernelGGL(hyp_attn_kernel, grid, block, SMEM_BYTES, stream,
                       qs, ksp, vsp, scale, bias, out);
}
